// HGNN_conv_631
// MI455X (gfx1250) — compile-verified, matrix operand order checked by reading, not run on hardware
//
#include <hip/hip_runtime.h>
#include <hip/hip_bf16.h>

typedef __attribute__((ext_vector_type(16))) _Float16 v16h;
typedef __attribute__((ext_vector_type(8)))  _Float16 v8h;
typedef __attribute__((ext_vector_type(4)))  _Float16 v4h;
typedef __attribute__((ext_vector_type(8)))  float    v8f;
typedef __attribute__((ext_vector_type(4)))  int      vi4;

#define N_NODES 20000
#define N_EDGES 4096
#define FT      128

#define BM 32          // rows per block
#define BN 128         // full feature dim -> H read exactly once per GEMM
#define BK 32
#define LDT 40         // padded LDS row stride in halves (80B, 16B-aligned)

#define KSPLIT   4
#define G1_ITERS (N_NODES / BK)                    // 625
#define G1_CHUNK ((G1_ITERS + KSPLIT - 1) / KSPLIT) // 157

#define DEG_BLOCKS 250
#define DEG_ROWS   80   // 250 * 80 = 20000

// ---------------------------------------------------------------------------
// WMMA fragment helpers (f16 16x16x32, CDNA5 ISA VGPR layouts).
// A (16x32): lane L holds row m=L%16; elems 0..7 = K=kb..kb+7,
// elems 8..15 = K=16+kb..16+kb+7, kb = 8*(L/16). B symmetric with n=L%16.
// Tiles staged in LDS as [row][k] -> fragment = two contiguous ds_load_b128.
// ---------------------------------------------------------------------------
__device__ __forceinline__ v16h load_frag(const _Float16* base, int lane) {
  const int r  = lane & 15;
  const int kb = (lane & 16) >> 1;  // 0 or 8
  const _Float16* p = base + r * LDT + kb;
  v8h lo = *(const v8h*)(p);
  v8h hi = *(const v8h*)(p + 16);
  return __builtin_shufflevector(lo, hi, 0,1,2,3,4,5,6,7,8,9,10,11,12,13,14,15);
}

__device__ __forceinline__ v8f wmma_f16(v16h a, v16h b, v8f c) {
  return __builtin_amdgcn_wmma_f32_16x16x32_f16(false, a, false, b,
                                                (short)0, c, false, false);
}

// ---------------------------------------------------------------------------
// Prep kernels
// ---------------------------------------------------------------------------
__global__ void zero_f32(float* p, int n) {
  int i = blockIdx.x * blockDim.x + threadIdx.x;
  if (i < n) p[i] = 0.0f;
}

// Single pass over H: row sums -> dv_inv_sqrt; column sums accumulated in LDS,
// flushed with float atomics (integer-valued -> bit-deterministic).
__global__ void __launch_bounds__(256) deg_kernel(const int* __restrict__ H,
                                                  float* __restrict__ dv_inv_sqrt,
                                                  float* __restrict__ e_acc) {
  __shared__ int   eloc[N_EDGES];   // 16 KB
  __shared__ float wsum[8];
  const int tid = threadIdx.x;
  for (int c = tid; c < N_EDGES; c += 256) eloc[c] = 0;
  __syncthreads();

  const int r0 = blockIdx.x * DEG_ROWS;
  for (int rr = 0; rr < DEG_ROWS; ++rr) {
    const int r = r0 + rr;
    const int* row = H + (size_t)r * N_EDGES;
    int s = 0;
    #pragma unroll
    for (int j = 0; j < N_EDGES / 256; ++j) {
      int v = row[tid + 256 * j];
      s += v;
      eloc[tid + 256 * j] += v;     // column owned by this thread
    }
    #pragma unroll
    for (int off = 16; off > 0; off >>= 1) s += __shfl_down(s, off, 32);
    if ((tid & 31) == 0) wsum[tid >> 5] = (float)s;
    __syncthreads();
    if (tid == 0) {
      float t = 0.0f;
      #pragma unroll
      for (int w = 0; w < 8; ++w) t += wsum[w];
      dv_inv_sqrt[r] = rsqrtf(fmaxf(t, 1.0f));
    }
    __syncthreads();
  }
  for (int c = tid; c < N_EDGES; c += 256)
    atomicAdd(&e_acc[c], (float)eloc[c]);
}

__global__ void finalize_e(const float* __restrict__ e_acc, float* __restrict__ de_inv) {
  int i = blockIdx.x * blockDim.x + threadIdx.x;
  if (i < N_EDGES) de_inv[i] = 1.0f / fmaxf(e_acc[i], 1.0f);
}

// Wh_t[n][k] = f16(weight[k][n])  (transposed so B staging is contiguous)
__global__ void convert_wt(const float* __restrict__ W, _Float16* __restrict__ Wht) {
  int i = blockIdx.x * blockDim.x + threadIdx.x;
  if (i < FT * FT) {
    int n = i >> 7, k = i & 127;
    Wht[i] = (_Float16)W[(size_t)k * FT + n];
  }
}

// Yh_t[f][i] = f16(dv[i] * X[i][f])  via LDS transpose, coalesced both sides.
__global__ void __launch_bounds__(256) make_yht(const float* __restrict__ X,
                                                const float* __restrict__ dv,
                                                _Float16* __restrict__ Yht) {
  __shared__ float xt[32][FT + 4];
  __shared__ float dvl[32];
  const int tid = threadIdx.x;
  const int i0 = blockIdx.x * 32;      // 625 blocks
  #pragma unroll
  for (int rep = 0; rep < 16; ++rep) {
    int idx = rep * 256 + tid;
    int r = idx >> 7, c = idx & 127;
    xt[r][c] = X[(size_t)(i0 + r) * FT + c];
  }
  if (tid < 32) dvl[tid] = dv[i0 + tid];
  __syncthreads();
  const int f = tid >> 1, jh = (tid & 1) * 16;
  v8h o0, o1;
  #pragma unroll
  for (int j = 0; j < 8; ++j) {
    o0[j] = (_Float16)(dvl[jh + j]     * xt[jh + j][f]);
    o1[j] = (_Float16)(dvl[jh + 8 + j] * xt[jh + 8 + j][f]);
  }
  *(v8h*)&Yht[(size_t)f * N_NODES + i0 + jh]     = o0;
  *(v8h*)&Yht[(size_t)f * N_NODES + i0 + jh + 8] = o1;
}

// ---------------------------------------------------------------------------
// GEMM1 (split-K): P[c][e][f] = sum_{k in chunk c} H[k][e] * Yh_t[f][k]
// M=4096(e), N=128(f), K=20000. grid (128, 4), 256 threads, 8 waves,
// wave tile 16x32 (wm = w&1, wn = w>>1).
// ---------------------------------------------------------------------------
__global__ void __launch_bounds__(256) gemm1_hT_y(const int* __restrict__ H,
                                                  const _Float16* __restrict__ Yht,
                                                  float* __restrict__ P) {
  __shared__ __align__(16) _Float16 ldsA[BM][LDT];
  __shared__ __align__(16) _Float16 ldsB[BN][LDT];
  const int tid = threadIdx.x, lane = tid & 31, wave = tid >> 5;
  const int wm = wave & 1, wn = wave >> 1;
  const int m0 = blockIdx.x * BM;            // hyperedge tile
  const int it0 = blockIdx.y * G1_CHUNK;
  int it1 = it0 + G1_CHUNK; if (it1 > G1_ITERS) it1 = G1_ITERS;

  const int am  = tid & 31;   // A: m within tile (lane-contiguous -> coalesced H)
  const int akq = tid >> 5;   // A: k quad (k = akq*4 + j)

  v8f acc0 = {}, acc1 = {};

  for (int it = it0; it < it1; ++it) {
    const int k0 = it * BK;
    // A = H^T tile (transpose while converting): ldsA[m][k] = f16(H[k0+k][m0+m])
    v4h ap;
    #pragma unroll
    for (int j = 0; j < 4; ++j)
      ap[j] = (_Float16)H[(size_t)(k0 + akq * 4 + j) * N_EDGES + (m0 + am)];
    *(v4h*)&ldsA[am][akq * 4] = ap;
    // B tile: ldsB[n][k] = Yh_t[n][k0+k]  (contiguous 16B copies)
    #pragma unroll
    for (int rep = 0; rep < 2; ++rep) {
      int idx = rep * 256 + tid;
      int kg = idx & 3, n = idx >> 2;
      *(v8h*)&ldsB[n][kg * 8] =
          *(const v8h*)&Yht[(size_t)n * N_NODES + k0 + kg * 8];
    }
    __syncthreads();
    v16h a  = load_frag(&ldsA[16 * wm][0], lane);
    v16h b0 = load_frag(&ldsB[32 * wn][0], lane);
    v16h b1 = load_frag(&ldsB[32 * wn + 16][0], lane);
    acc0 = wmma_f16(a, b0, acc0);
    acc1 = wmma_f16(a, b1, acc1);
    __syncthreads();
  }

  const int mloc = (lane >> 4) << 3, nloc = lane & 15;
  float* Pc = P + (size_t)blockIdx.y * N_EDGES * FT;
  #pragma unroll
  for (int i = 0; i < 8; ++i) {
    int e = m0 + 16 * wm + mloc + i;
    int f = 32 * wn + nloc;
    Pc[(size_t)e * FT + f]      = acc0[i];
    Pc[(size_t)e * FT + f + 16] = acc1[i];
  }
}

// Reduce split-K partials, scale by de_inv, emit transposed f16 Th_t[f][e].
__global__ void __launch_bounds__(256) reduce1(const float* __restrict__ P,
                                               const float* __restrict__ de_inv,
                                               _Float16* __restrict__ Tht) {
  __shared__ float t[32][FT + 4];
  const int tid = threadIdx.x;
  const int e0 = blockIdx.x * 32;            // 128 blocks
  #pragma unroll
  for (int rep = 0; rep < 16; ++rep) {
    int idx = rep * 256 + tid;
    int er = idx >> 7, f = idx & 127;
    size_t o = (size_t)(e0 + er) * FT + f;
    const size_t st = (size_t)N_EDGES * FT;
    t[er][f] = P[o] + P[o + st] + P[o + 2 * st] + P[o + 3 * st];
  }
  __syncthreads();
  const int f = tid >> 1, eh = (tid & 1) * 16;
  v8h o0, o1;
  #pragma unroll
  for (int j = 0; j < 8; ++j) {
    o0[j] = (_Float16)(de_inv[e0 + eh + j]     * t[eh + j][f]);
    o1[j] = (_Float16)(de_inv[e0 + eh + 8 + j] * t[eh + 8 + j][f]);
  }
  *(v8h*)&Tht[(size_t)f * N_EDGES + e0 + eh]     = o0;
  *(v8h*)&Tht[(size_t)f * N_EDGES + e0 + eh + 8] = o1;
}

// ---------------------------------------------------------------------------
// GEMM2: Zs[i][f] = f16( dv[i] * sum_e H[i][e] * Th_t[f][e] )
// M=20000, N=128, K=4096. grid 625 (exact, no edge guards).
// ---------------------------------------------------------------------------
__global__ void __launch_bounds__(256) gemm2_h_t(const int* __restrict__ H,
                                                 const _Float16* __restrict__ Tht,
                                                 const float* __restrict__ dv,
                                                 _Float16* __restrict__ Zs) {
  __shared__ __align__(16) _Float16 ldsA[BM][LDT];
  __shared__ __align__(16) _Float16 ldsB[BN][LDT];
  const int tid = threadIdx.x, lane = tid & 31, wave = tid >> 5;
  const int wm = wave & 1, wn = wave >> 1;
  const int m0 = blockIdx.x * BM;            // node tile
  const int am  = tid >> 3;   // A: row (4 rows per wave)
  const int akq = tid & 7;    // A: k quad -> one global_load_b128 of H

  v8f acc0 = {}, acc1 = {};

  for (int k0 = 0; k0 < N_EDGES; k0 += BK) {
    vi4 hv = *(const vi4*)&H[(size_t)(m0 + am) * N_EDGES + k0 + akq * 4];
    v4h ap;
    #pragma unroll
    for (int j = 0; j < 4; ++j) ap[j] = (_Float16)hv[j];
    *(v4h*)&ldsA[am][akq * 4] = ap;
    #pragma unroll
    for (int rep = 0; rep < 2; ++rep) {
      int idx = rep * 256 + tid;
      int kg = idx & 3, n = idx >> 2;
      *(v8h*)&ldsB[n][kg * 8] =
          *(const v8h*)&Tht[(size_t)n * N_EDGES + k0 + kg * 8];
    }
    __syncthreads();
    v16h a  = load_frag(&ldsA[16 * wm][0], lane);
    v16h b0 = load_frag(&ldsB[32 * wn][0], lane);
    v16h b1 = load_frag(&ldsB[32 * wn + 16][0], lane);
    acc0 = wmma_f16(a, b0, acc0);
    acc1 = wmma_f16(a, b1, acc1);
    __syncthreads();
  }

  const int mloc = (lane >> 4) << 3, nloc = lane & 15;
  #pragma unroll
  for (int i = 0; i < 8; ++i) {
    int m = m0 + 16 * wm + mloc + i;
    int f = 32 * wn + nloc;
    float s = dv[m];
    Zs[(size_t)m * FT + f]      = (_Float16)(s * acc0[i]);
    Zs[(size_t)m * FT + f + 16] = (_Float16)(s * acc1[i]);
  }
}

// ---------------------------------------------------------------------------
// GEMM3: out[i][f] = sum_k Zs[i][k] * Wh_t[f][k] + bias[f]   (f32 out)
// M=20000, N=128, K=128. grid 625.
// ---------------------------------------------------------------------------
__global__ void __launch_bounds__(256) gemm3_out(const _Float16* __restrict__ Zs,
                                                 const _Float16* __restrict__ Wht,
                                                 const float* __restrict__ bias,
                                                 float* __restrict__ out) {
  __shared__ __align__(16) _Float16 ldsA[BM][LDT];
  __shared__ __align__(16) _Float16 ldsB[BN][LDT];
  const int tid = threadIdx.x, lane = tid & 31, wave = tid >> 5;
  const int wm = wave & 1, wn = wave >> 1;
  const int m0 = blockIdx.x * BM;
  const int am  = tid >> 3;
  const int akq = tid & 7;

  v8f acc0 = {}, acc1 = {};

  for (int k0 = 0; k0 < FT; k0 += BK) {   // 4 iterations
    *(v4h*)&ldsA[am][akq * 4] =
        *(const v4h*)&Zs[(size_t)(m0 + am) * FT + k0 + akq * 4];
    #pragma unroll
    for (int rep = 0; rep < 2; ++rep) {
      int idx = rep * 256 + tid;
      int kg = idx & 3, n = idx >> 2;
      *(v8h*)&ldsB[n][kg * 8] =
          *(const v8h*)&Wht[(size_t)n * FT + k0 + kg * 8];
    }
    __syncthreads();
    v16h a  = load_frag(&ldsA[16 * wm][0], lane);
    v16h b0 = load_frag(&ldsB[32 * wn][0], lane);
    v16h b1 = load_frag(&ldsB[32 * wn + 16][0], lane);
    acc0 = wmma_f16(a, b0, acc0);
    acc1 = wmma_f16(a, b1, acc1);
    __syncthreads();
  }

  const int mloc = (lane >> 4) << 3, nloc = lane & 15;
  #pragma unroll
  for (int i = 0; i < 8; ++i) {
    int m = m0 + 16 * wm + mloc + i;
    int f = 32 * wn + nloc;
    out[(size_t)m * FT + f]      = acc0[i] + bias[f];
    out[(size_t)m * FT + f + 16] = acc1[i] + bias[f + 16];
  }
}

// ---------------------------------------------------------------------------
// Launch
// ---------------------------------------------------------------------------
extern "C" void kernel_launch(void* const* d_in, const int* in_sizes, int n_in,
                              void* d_out, int out_size, void* d_ws, size_t ws_size,
                              hipStream_t stream) {
  const float* X      = (const float*)d_in[0];
  const int*   H      = (const int*)d_in[1];
  const float* weight = (const float*)d_in[2];
  const float* bias   = (const float*)d_in[3];
  float* out = (float*)d_out;

  char* ws = (char*)d_ws;
  size_t off = 0;
  auto carve = [&](size_t bytes) -> char* {
    char* p = ws + off;
    off += (bytes + 255) & ~(size_t)255;
    return p;
  };
  float*    dv     = (float*)   carve((size_t)N_NODES * 4);
  float*    e_acc  = (float*)   carve((size_t)N_EDGES * 4);
  float*    de_inv = (float*)   carve((size_t)N_EDGES * 4);
  _Float16* Yht    = (_Float16*)carve((size_t)FT * N_NODES * 2);
  _Float16* Tht    = (_Float16*)carve((size_t)FT * N_EDGES * 2);
  _Float16* Zs     = (_Float16*)carve((size_t)N_NODES * FT * 2);
  _Float16* Wht    = (_Float16*)carve((size_t)FT * FT * 2);
  float*    P      = (float*)   carve((size_t)KSPLIT * N_EDGES * FT * 4);
  (void)ws_size; (void)in_sizes; (void)n_in; (void)out_size;

  zero_f32<<<(N_EDGES + 255) / 256, 256, 0, stream>>>(e_acc, N_EDGES);
  deg_kernel<<<DEG_BLOCKS, 256, 0, stream>>>(H, dv, e_acc);
  finalize_e<<<(N_EDGES + 255) / 256, 256, 0, stream>>>(e_acc, de_inv);
  convert_wt<<<(FT * FT + 255) / 256, 256, 0, stream>>>(weight, Wht);
  make_yht<<<N_NODES / 32, 256, 0, stream>>>(X, dv, Yht);

  gemm1_hT_y<<<dim3(N_EDGES / BM, KSPLIT), 256, 0, stream>>>(H, Yht, P);
  reduce1<<<N_EDGES / 32, 256, 0, stream>>>(P, de_inv, Tht);
  gemm2_h_t<<<N_NODES / BM, 256, 0, stream>>>(H, Tht, dv, Zs);
  gemm3_out<<<N_NODES / BM, 256, 0, stream>>>(Zs, Wht, bias, out);
}
